// MultiHeadAttention_86552180949396
// MI455X (gfx1250) — matrix operand order repaired and checked by reading, not compiled, not run on hardware
//
#include <hip/hip_runtime.h>
#include <hip/hip_bf16.h>
#include <stdint.h>

#define SEQ   2048
#define EMB   1024
#define HEADS 16
#define HD    64
#define MTOT  4096   // B*L

typedef __attribute__((ext_vector_type(16))) _Float16 v16h;
typedef __attribute__((ext_vector_type(8)))  float    v8f;
typedef __attribute__((ext_vector_type(4)))  int      v4i;

union HFrag {
    v16h     v;
    _Float16 h[16];
    uint32_t u[8];
};

// CDNA5 WMMA 16-bit A-matrix (16x32) K-base per VGPR v, lane-group g
__device__ __forceinline__ int a_kbase(int v, int g) {
    return (v < 4) ? (8 * g + 2 * v) : (16 + 8 * g + 2 * (v - 4));
}
// CDNA5 WMMA 16-bit B-matrix (32x16) K-base per VGPR v, lane-group g
__device__ __forceinline__ int b_kbase(int v, int g) {
    return (v < 4) ? (8 * g + 2 * v) : (16 + 8 * g + 2 * (v - 4));
}

// ---------------- async global->LDS copy (ASYNCcnt path), guarded -----------
#if defined(__has_builtin)
#if __has_builtin(__builtin_amdgcn_global_load_async_to_lds_b128)
#define HAS_ASYNC_LDS 1
#endif
#endif

#ifdef HAS_ASYNC_LDS
typedef __attribute__((address_space(1))) v4i as1_v4i;
typedef __attribute__((address_space(3))) v4i as3_v4i;
#endif

// copy 16 bytes global -> LDS (per active lane)
__device__ __forceinline__ void copy16_g2l(const void* g, void* l) {
#ifdef HAS_ASYNC_LDS
    __builtin_amdgcn_global_load_async_to_lds_b128(
        (as1_v4i*)(uintptr_t)g, (as3_v4i*)(uint32_t)(uintptr_t)l, 0, 0);
#else
    *(uint4*)l = *(const uint4*)g;
#endif
}
__device__ __forceinline__ void copy_wait() {
#ifdef HAS_ASYNC_LDS
#if __has_builtin(__builtin_amdgcn_s_wait_asynccnt)
    __builtin_amdgcn_s_wait_asynccnt(0);
#else
    asm volatile("s_wait_asynccnt 0" ::: "memory");
#endif
#endif
}

// ---------------------------------------------------------------- convert ---
__global__ __launch_bounds__(256) void f32_to_f16_kernel(const float* __restrict__ in,
                                                         _Float16* __restrict__ out, int n) {
    for (int i = blockIdx.x * blockDim.x + threadIdx.x; i < n; i += gridDim.x * blockDim.x)
        out[i] = (_Float16)in[i];
}

// -------------------------------------------------- shared GEMM tile load ---
// 128x64 f16 tiles of X and W (both K-contiguous row-major), LDS row stride 72
__device__ __forceinline__ void load_gemm_tile(const _Float16* __restrict__ X,
                                               const _Float16* __restrict__ W,
                                               _Float16* sA, _Float16* sB,
                                               int tid, int m0, int n0, int k0) {
#pragma unroll
    for (int c = 0; c < 4; ++c) {
        int chunk = tid + c * 256;                 // 1024 chunks of 8 halves
        int r = chunk >> 3, col = (chunk & 7) * 8;
        copy16_g2l(&X[(size_t)(m0 + r) * EMB + k0 + col], &sA[r * 72 + col]);
        copy16_g2l(&W[(size_t)(n0 + r) * EMB + k0 + col], &sB[r * 72 + col]);
    }
}

// ------------------------------------------------------------- QKV GEMM -----
// C(4096x3072) = X(4096x1024) * W_in^T ; scatter into Q(scaled),K,V head-major
__global__ __launch_bounds__(256) void gemm_qkv_kernel(
    const _Float16* __restrict__ X, const _Float16* __restrict__ W,
    _Float16* __restrict__ Qh, _Float16* __restrict__ Kh, _Float16* __restrict__ Vh)
{
    __shared__ _Float16 sA[2][128 * 72];
    __shared__ _Float16 sB[2][128 * 72];
    const int tid  = threadIdx.x;
    const int lane = tid & 31, wid = tid >> 5;
    const int wm   = wid >> 1, wn = wid & 1;
    const int g    = lane >> 4, ln = lane & 15;
    const int m0   = blockIdx.y * 128, n0 = blockIdx.x * 128;

    v8f acc[2][4] = {};

    load_gemm_tile(X, W, sA[0], sB[0], tid, m0, n0, 0);
    for (int k0 = 0; k0 < EMB; k0 += 64) {
        const int buf = (k0 >> 6) & 1;
        copy_wait();          // this wave's async copies for tile `buf` done
        __syncthreads();      // everyone's copies done + prev compute finished
        if (k0 + 64 < EMB)
            load_gemm_tile(X, W, sA[buf ^ 1], sB[buf ^ 1], tid, m0, n0, k0 + 64);

#pragma unroll
        for (int kk = 0; kk < 2; ++kk) {
            HFrag a[2], b[4];
#pragma unroll
            for (int i = 0; i < 2; ++i) {
                const _Float16* base = &sA[buf][(wm * 32 + i * 16 + ln) * 72 + kk * 32];
#pragma unroll
                for (int v = 0; v < 8; ++v) a[i].u[v] = *(const uint32_t*)&base[a_kbase(v, g)];
            }
#pragma unroll
            for (int j = 0; j < 4; ++j) {
                const _Float16* base = &sB[buf][(wn * 64 + j * 16 + ln) * 72 + kk * 32];
#pragma unroll
                for (int v = 0; v < 8; ++v) b[j].u[v] = *(const uint32_t*)&base[b_kbase(v, g)];
            }
#pragma unroll
            for (int i = 0; i < 2; ++i)
#pragma unroll
                for (int j = 0; j < 4; ++j)
                    acc[i][j] = __builtin_amdgcn_wmma_f32_16x16x32_f16(
                        false, a[i].v, false, b[j].v, (short)0, acc[i][j], false, false);
        }
    }

#pragma unroll
    for (int i = 0; i < 2; ++i)
#pragma unroll
        for (int j = 0; j < 4; ++j)
#pragma unroll
            for (int r = 0; r < 8; ++r) {
                int   row = m0 + wm * 32 + i * 16 + g * 8 + r;
                int   col = n0 + wn * 64 + j * 16 + ln;
                float val = acc[i][j][r];
                int which = col >> 10, e = col & 1023;
                int head = e >> 6, d = e & 63;
                int bb = row >> 11, l = row & 2047;
                size_t idx = (((size_t)(bb * HEADS + head)) * SEQ + l) * HD + d;
                if (which == 0)      Qh[idx] = (_Float16)(val * 0.125f);   // 1/sqrt(64)
                else if (which == 1) Kh[idx] = (_Float16)val;
                else                 Vh[idx] = (_Float16)val;
            }
}

// --------------------------------------------------- flash attention ---------
// grid: (L/128, B*H). Each wave owns 16 q rows; streams 32-row K/V tiles.
__global__ __launch_bounds__(256) void attn_alibi_kernel(
    const _Float16* __restrict__ Qh, const _Float16* __restrict__ Kh,
    const _Float16* __restrict__ Vh, _Float16* __restrict__ Ctx)
{
    __shared__ _Float16 sK[32 * 72];      // K tile, row-major (kk, d)
    __shared__ _Float16 sVt[64 * 40];     // V tile transposed (d, kk)
    __shared__ _Float16 sP[8][16 * 40];   // per-wave P scratch (16x32, stride 40)

    const int tid  = threadIdx.x;
    const int lane = tid & 31, wid = tid >> 5;
    const int g    = lane >> 4, ln = lane & 15;
    const int bh   = blockIdx.y;
    const int head = bh & (HEADS - 1);
    const float slope = exp2f(-0.5f * (float)(head + 1));   // 2^(-8/H*(h+1))
    const int q0 = blockIdx.x * 128;
    const int qw = q0 + wid * 16;
    const size_t base_bh = (size_t)bh * SEQ;

    // Q fragments: K-dim 64 as two 32-chunks
    HFrag aQ[2];
    {
        const _Float16* qrow = &Qh[(base_bh + qw + ln) * HD];
#pragma unroll
        for (int c = 0; c < 2; ++c)
#pragma unroll
            for (int v = 0; v < 8; ++v)
                aQ[c].u[v] = *(const uint32_t*)&qrow[c * 32 + a_kbase(v, g)];
    }

    float mrun[8], lrun[8];
    v8f   o[4] = {};
#pragma unroll
    for (int r = 0; r < 8; ++r) { mrun[r] = -1e30f; lrun[r] = 0.0f; }

    for (int k0 = 0; k0 < q0 + 128; k0 += 32) {
        __syncthreads();
        {   // K tile via async copy; V tile loaded + transposed manually
            int r = tid >> 3, col = (tid & 7) * 8;
            copy16_g2l(&Kh[(base_bh + k0 + r) * HD + col], &sK[r * 72 + col]);
            HFrag tv;
            *(uint4*)&tv.h[0] = *(const uint4*)&Vh[(base_bh + k0 + r) * HD + col];
#pragma unroll
            for (int t = 0; t < 8; ++t) sVt[(col + t) * 40 + r] = tv.h[t];
        }
        copy_wait();
        __syncthreads();

        // scores S(16x32) = Q(16x64) * K_tile^T
        v8f s[2] = {};
#pragma unroll
        for (int j = 0; j < 2; ++j)
#pragma unroll
            for (int c = 0; c < 2; ++c) {
                HFrag bk;
                const _Float16* kb = &sK[(j * 16 + ln) * 72 + c * 32];
#pragma unroll
                for (int v = 0; v < 8; ++v) bk.u[v] = *(const uint32_t*)&kb[b_kbase(v, g)];
                s[j] = __builtin_amdgcn_wmma_f32_16x16x32_f16(
                    false, aQ[c].v, false, bk.v, (short)0, s[j], false, false);
            }

        // ALiBi bias + causal mask in C layout (row = r+8g, col = j*16+ln)
        float p[2][8];
#pragma unroll
        for (int r = 0; r < 8; ++r) {
            int i = qw + g * 8 + r;
#pragma unroll
            for (int j = 0; j < 2; ++j) {
                int   col = k0 + j * 16 + ln;
                float sv  = s[j][r];
                p[j][r]   = (col > i) ? -1e30f : (sv + slope * (float)(col - i));
            }
        }
        // online softmax: reduce across the 16-lane N group
#pragma unroll
        for (int r = 0; r < 8; ++r) {
            float mx = fmaxf(p[0][r], p[1][r]);
#pragma unroll
            for (int m = 1; m < 16; m <<= 1) mx = fmaxf(mx, __shfl_xor(mx, m, 32));
            float mnew = fmaxf(mrun[r], mx);
            float sc   = __expf(mrun[r] - mnew);
            mrun[r]    = mnew;
            float e0 = __expf(p[0][r] - mnew);
            float e1 = __expf(p[1][r] - mnew);
            p[0][r] = e0; p[1][r] = e1;
            float sum = e0 + e1;
#pragma unroll
            for (int m = 1; m < 16; m <<= 1) sum += __shfl_xor(sum, m, 32);
            lrun[r] = lrun[r] * sc + sum;
#pragma unroll
            for (int t = 0; t < 4; ++t) o[t][r] *= sc;
        }
        // bounce P through per-wave LDS: C layout -> A fragment layout
        _Float16* sPw = sP[wid];
#pragma unroll
        for (int r = 0; r < 8; ++r) {
            int lr = g * 8 + r;
            sPw[lr * 40 + ln]      = (_Float16)p[0][r];
            sPw[lr * 40 + 16 + ln] = (_Float16)p[1][r];
        }
        HFrag pa;
        {
            const _Float16* base = &sPw[ln * 40];
#pragma unroll
            for (int v = 0; v < 8; ++v) pa.u[v] = *(const uint32_t*)&base[a_kbase(v, g)];
        }
        // O(16x64) += P(16x32) * V(32x64)
#pragma unroll
        for (int t = 0; t < 4; ++t) {
            HFrag bv;
            const _Float16* vb = &sVt[(t * 16 + ln) * 40];
#pragma unroll
            for (int v = 0; v < 8; ++v) bv.u[v] = *(const uint32_t*)&vb[b_kbase(v, g)];
            o[t] = __builtin_amdgcn_wmma_f32_16x16x32_f16(
                false, pa.v, false, bv.v, (short)0, o[t], false, false);
        }
    }

    // normalize and write ctx as f16 (B, L, E) with e = head*64 + d
    const int bb = bh >> 4;
#pragma unroll
    for (int r = 0; r < 8; ++r) {
        float inv = 1.0f / lrun[r];
        int   l   = qw + g * 8 + r;
        size_t rowbase = ((size_t)(bb * SEQ + l)) * EMB + head * HD;
#pragma unroll
        for (int t = 0; t < 4; ++t)
            Ctx[rowbase + t * 16 + ln] = (_Float16)(o[t][r] * inv);
    }
}

// ------------------------------------------------------------- out GEMM -----
// Out(4096x1024) = Ctx(4096x1024) * W_out^T, f32 output
__global__ __launch_bounds__(256) void gemm_out_kernel(
    const _Float16* __restrict__ X, const _Float16* __restrict__ W, float* __restrict__ Out)
{
    __shared__ _Float16 sA[2][128 * 72];
    __shared__ _Float16 sB[2][128 * 72];
    const int tid  = threadIdx.x;
    const int lane = tid & 31, wid = tid >> 5;
    const int wm   = wid >> 1, wn = wid & 1;
    const int g    = lane >> 4, ln = lane & 15;
    const int m0   = blockIdx.y * 128, n0 = blockIdx.x * 128;

    v8f acc[2][4] = {};

    load_gemm_tile(X, W, sA[0], sB[0], tid, m0, n0, 0);
    for (int k0 = 0; k0 < EMB; k0 += 64) {
        const int buf = (k0 >> 6) & 1;
        copy_wait();
        __syncthreads();
        if (k0 + 64 < EMB)
            load_gemm_tile(X, W, sA[buf ^ 1], sB[buf ^ 1], tid, m0, n0, k0 + 64);

#pragma unroll
        for (int kk = 0; kk < 2; ++kk) {
            HFrag a[2], b[4];
#pragma unroll
            for (int i = 0; i < 2; ++i) {
                const _Float16* base = &sA[buf][(wm * 32 + i * 16 + ln) * 72 + kk * 32];
#pragma unroll
                for (int v = 0; v < 8; ++v) a[i].u[v] = *(const uint32_t*)&base[a_kbase(v, g)];
            }
#pragma unroll
            for (int j = 0; j < 4; ++j) {
                const _Float16* base = &sB[buf][(wn * 64 + j * 16 + ln) * 72 + kk * 32];
#pragma unroll
                for (int v = 0; v < 8; ++v) b[j].u[v] = *(const uint32_t*)&base[b_kbase(v, g)];
            }
#pragma unroll
            for (int i = 0; i < 2; ++i)
#pragma unroll
                for (int j = 0; j < 4; ++j)
                    acc[i][j] = __builtin_amdgcn_wmma_f32_16x16x32_f16(
                        false, a[i].v, false, b[j].v, (short)0, acc[i][j], false, false);
        }
    }

#pragma unroll
    for (int i = 0; i < 2; ++i)
#pragma unroll
        for (int j = 0; j < 4; ++j)
#pragma unroll
            for (int r = 0; r < 8; ++r) {
                int row = m0 + wm * 32 + i * 16 + g * 8 + r;
                int col = n0 + wn * 64 + j * 16 + ln;
                Out[(size_t)row * EMB + col] = acc[i][j][r];
            }
}

// ---------------------------------------------------------------- launch ----
extern "C" void kernel_launch(void* const* d_in, const int* in_sizes, int n_in,
                              void* d_out, int out_size, void* d_ws, size_t ws_size,
                              hipStream_t stream) {
    const float* x     = (const float*)d_in[0];   // (2, 2048, 1024)
    const float* w_in  = (const float*)d_in[1];   // (3072, 1024)
    const float* w_out = (const float*)d_in[2];   // (1024, 1024)
    float*       out   = (float*)d_out;           // (2, 2048, 1024)

    char* ws = (char*)d_ws;
    const size_t MB = 1u << 20;
    _Float16* xh    = (_Float16*)(ws);               // 8 MiB  (reused as ctx)
    _Float16* ctxh  = xh;
    _Float16* winh  = (_Float16*)(ws + 8 * MB);      // 6 MiB
    _Float16* wouth = (_Float16*)(ws + 14 * MB);     // 2 MiB
    _Float16* Qh    = (_Float16*)(ws + 16 * MB);     // 8 MiB
    _Float16* Kh    = (_Float16*)(ws + 24 * MB);     // 8 MiB
    _Float16* Vh    = (_Float16*)(ws + 32 * MB);     // 8 MiB  (total 40 MiB)

    f32_to_f16_kernel<<<1024, 256, 0, stream>>>(x,     xh,    MTOT * EMB);
    f32_to_f16_kernel<<<1024, 256, 0, stream>>>(w_in,  winh,  3 * EMB * EMB);
    f32_to_f16_kernel<<<1024, 256, 0, stream>>>(w_out, wouth, EMB * EMB);

    gemm_qkv_kernel<<<dim3(24, 32), 256, 0, stream>>>(xh, winh, Qh, Kh, Vh);
    attn_alibi_kernel<<<dim3(SEQ / 128, 2 * HEADS), 256, 0, stream>>>(Qh, Kh, Vh, ctxh);
    gemm_out_kernel<<<dim3(8, 32), 256, 0, stream>>>(ctxh, wouth, out);
}
